// Head_1288490189201
// MI455X (gfx1250) — matrix operand order repaired and checked by reading, not compiled, not run on hardware
//
#include <hip/hip_runtime.h>
#include <hip/hip_bf16.h>
#include <math.h>

typedef __bf16 bf16;
typedef __attribute__((ext_vector_type(16))) __bf16 v16bf;
typedef __attribute__((ext_vector_type(8)))  __bf16 bf16x8;
typedef __attribute__((ext_vector_type(8)))  float  v8f;

#define N_EMBD 1024
#define BATCH  2
#define TLEN   4096
#define ROWS   (BATCH * TLEN)   // 8192

union FragU { v16bf v; bf16x8 h[2]; };

// A-matrix 16x32 bf16 (MxK), ISA 7.12.2: lane&15 = M row; lane>>4 picks K-halves.
__device__ __forceinline__ v16bf load_a_frag(const bf16* base, int lane, int rowStride) {
  const int m = lane & 15, kh = lane >> 4;
  const bf16* p = base + (size_t)m * rowStride + kh * 8;
  FragU f;
  f.h[0] = *(const bf16x8*)(p);
  f.h[1] = *(const bf16x8*)(p + 16);
  return f.v;
}

// B-matrix 32x16 bf16 (KxN): lane&15 = N column; lane>>4 picks K-halves (same K order as A).
__device__ __forceinline__ v16bf load_b_frag(const bf16* base, int lane, int rowStride) {
  const int n = lane & 15, kh = lane >> 4;
  const bf16* p = base + (size_t)(kh * 8) * rowStride + n;
  FragU f;
#pragma unroll
  for (int s = 0; s < 8; ++s) {
    f.v[s]     = p[(size_t)s * rowStride];
    f.v[8 + s] = p[(size_t)(16 + s) * rowStride];
  }
  return f.v;
}

__device__ __forceinline__ v8f wmma_bf16(v16bf a, v16bf b, v8f c) {
  return __builtin_amdgcn_wmma_f32_16x16x32_bf16(false, a, false, b, (short)0, c, false, false);
}

// ---------------------------------------------------------------------------
// Prep 1: f32 -> bf16 (x). 8 elements/thread.
// ---------------------------------------------------------------------------
__global__ void __launch_bounds__(256)
cvt_f32_bf16(const float* __restrict__ src, bf16* __restrict__ dst) {
  const size_t i = ((size_t)blockIdx.x * 256 + threadIdx.x) * 8;
  float4 a = *(const float4*)(src + i);
  float4 b = *(const float4*)(src + i + 4);
  bf16x8 o;
  o[0]=(bf16)a.x; o[1]=(bf16)a.y; o[2]=(bf16)a.z; o[3]=(bf16)a.w;
  o[4]=(bf16)b.x; o[5]=(bf16)b.y; o[6]=(bf16)b.z; o[7]=(bf16)b.w;
  *(bf16x8*)(dst + i) = o;
}

// ---------------------------------------------------------------------------
// Prep 2: Wt[c][o] = (bf16) W[o][c].  LDS-tiled 64x64 transpose+convert.
// ---------------------------------------------------------------------------
__global__ void __launch_bounds__(256)
transpose_cvt_wT(const float* __restrict__ W, bf16* __restrict__ Wt) {
  __shared__ bf16 tile[64][72];
  const int oBase = blockIdx.x * 64;
  const int cBase = blockIdx.y * 64;
  const int r = threadIdx.x >> 2, ch = threadIdx.x & 3;
  const float* p = W + (size_t)(oBase + r) * N_EMBD + cBase + ch * 16;
#pragma unroll
  for (int j = 0; j < 16; j += 4) {
    float4 v = *(const float4*)(p + j);
    tile[ch * 16 + j + 0][r] = (bf16)v.x;
    tile[ch * 16 + j + 1][r] = (bf16)v.y;
    tile[ch * 16 + j + 2][r] = (bf16)v.z;
    tile[ch * 16 + j + 3][r] = (bf16)v.w;
  }
  __syncthreads();
  bf16* q = Wt + (size_t)(cBase + r) * N_EMBD + oBase + ch * 16;
  bf16x8 o0, o1;
#pragma unroll
  for (int j = 0; j < 8; ++j) { o0[j] = tile[r][ch * 16 + j]; o1[j] = tile[r][ch * 16 + 8 + j]; }
  *(bf16x8*)(q)     = o0;
  *(bf16x8*)(q + 8) = o1;
}

// ---------------------------------------------------------------------------
// QKV GEMM: Y[m,n] = sum_c A[m,c] * Bt[c,n]   (all bf16, f32 accum)
// Block = 128x128 (8 waves), wave = 32x64 = 8 WMMA tiles. No LDS, no barriers.
// transposedStore: write Y as [batch][N_EMBD][TLEN] (used for K^T).
// ---------------------------------------------------------------------------
__global__ void __launch_bounds__(256)
gemm_bf16_nt(const bf16* __restrict__ A, const bf16* __restrict__ Bt,
             bf16* __restrict__ Y, float scale, int transposedStore) {
  const int lane = threadIdx.x & 31, w = threadIdx.x >> 5;
  const int mBase = blockIdx.x * 128 + (w & 3) * 32;
  const int nBase = blockIdx.y * 128 + (w >> 2) * 64;
  v8f acc[2][4] = {};
  for (int k = 0; k < N_EMBD; k += 32) {
    v16bf a[2], b[4];
    a[0] = load_a_frag(A + (size_t)(mBase)      * N_EMBD + k, lane, N_EMBD);
    a[1] = load_a_frag(A + (size_t)(mBase + 16) * N_EMBD + k, lane, N_EMBD);
#pragma unroll
    for (int j = 0; j < 4; ++j)
      b[j] = load_b_frag(Bt + (size_t)k * N_EMBD + nBase + 16 * j, lane, N_EMBD);
#pragma unroll
    for (int r = 0; r < 2; ++r)
#pragma unroll
      for (int j = 0; j < 4; ++j)
        acc[r][j] = wmma_bf16(a[r], b[j], acc[r][j]);
  }
  const int nc = lane & 15, mh = (lane >> 4) << 3;
#pragma unroll
  for (int r = 0; r < 2; ++r)
#pragma unroll
    for (int j = 0; j < 4; ++j)
#pragma unroll
      for (int i = 0; i < 8; ++i) {
        const int m = mBase + 16 * r + mh + i;
        const int n = nBase + 16 * j + nc;
        const bf16 v = (bf16)(acc[r][j][i] * scale);
        if (!transposedStore) {
          Y[(size_t)m * N_EMBD + n] = v;
        } else {                         // K^T: [batch][channel][token]
          const int bb = m >> 12, t = m & (TLEN - 1);
          Y[((size_t)bb * N_EMBD + n) * TLEN + t] = v;
        }
      }
}

// ---------------------------------------------------------------------------
// S = Q @ K^T, causal mask, bf16 out. Lower-triangular 128x128 tiles only.
// Pure global-fragment WMMA loop (K already transposed in memory).
// ---------------------------------------------------------------------------
__global__ void __launch_bounds__(256)
attn_scores(const bf16* __restrict__ Q, const bf16* __restrict__ Kt,
            bf16* __restrict__ P) {
  const int qTile = blockIdx.x, kTile = blockIdx.y, bz = blockIdx.z;
  if (kTile > qTile) return;
  const int lane = threadIdx.x & 31, w = threadIdx.x >> 5;
  const int mBase = qTile * 128 + (w & 3) * 32;
  const int nBase = kTile * 128 + (w >> 2) * 64;
  const bf16* Qb = Q  + (size_t)bz * TLEN * N_EMBD;
  const bf16* Kb = Kt + (size_t)bz * N_EMBD * TLEN;
  v8f acc[2][4] = {};
  for (int k = 0; k < N_EMBD; k += 32) {
    v16bf a[2], b[4];
    a[0] = load_a_frag(Qb + (size_t)(mBase)      * N_EMBD + k, lane, N_EMBD);
    a[1] = load_a_frag(Qb + (size_t)(mBase + 16) * N_EMBD + k, lane, N_EMBD);
#pragma unroll
    for (int j = 0; j < 4; ++j)
      b[j] = load_b_frag(Kb + (size_t)k * TLEN + nBase + 16 * j, lane, TLEN);
#pragma unroll
    for (int r = 0; r < 2; ++r)
#pragma unroll
      for (int j = 0; j < 4; ++j)
        acc[r][j] = wmma_bf16(a[r], b[j], acc[r][j]);
  }
  bf16* Pb = P + (size_t)bz * TLEN * TLEN;
  const int nc = lane & 15, mh = (lane >> 4) << 3;
#pragma unroll
  for (int r = 0; r < 2; ++r)
#pragma unroll
    for (int j = 0; j < 4; ++j)
#pragma unroll
      for (int i = 0; i < 8; ++i) {
        const int q = mBase + 16 * r + mh + i;
        const int n = nBase + 16 * j + nc;
        const float v = (n > q) ? -INFINITY : acc[r][j][i];
        Pb[(size_t)q * TLEN + n] = (bf16)v;
      }
}

// ---------------------------------------------------------------------------
// Row softmax in place (one wave32 per row, online max/sum).
// Bound = end of the row's 128-wide diagonal block (covers masked -inf region).
// ---------------------------------------------------------------------------
__global__ void __launch_bounds__(256)
softmax_rows(bf16* __restrict__ P) {
  const int lane = threadIdx.x & 31, w = threadIdx.x >> 5;
  const int r = blockIdx.x * 8 + w;
  const int bz = r >> 12, t = r & (TLEN - 1);
  bf16* rowp = P + (size_t)bz * TLEN * TLEN + (size_t)t * TLEN;
  const int nPos = ((((t >> 7) + 1) << 7) >> 3);  // # of 8-elem chunks
  float m = -1e30f, s = 0.f;
  for (int ic = lane; ic < nPos; ic += 32) {
    bf16x8 v = *(const bf16x8*)(rowp + ic * 8);
#pragma unroll
    for (int j = 0; j < 8; ++j) {
      float x  = (float)v[j];
      float nm = fmaxf(m, x);
      s = s * __expf(m - nm) + __expf(x - nm);
      m = nm;
    }
  }
#pragma unroll
  for (int off = 16; off > 0; off >>= 1) {
    float om = __shfl_xor(m, off, 32);
    float os = __shfl_xor(s, off, 32);
    float nm = fmaxf(m, om);
    s = s * __expf(m - nm) + os * __expf(om - nm);
    m = nm;
  }
  const float inv = 1.0f / s;
  for (int ic = lane; ic < nPos; ic += 32) {
    bf16x8 v = *(const bf16x8*)(rowp + ic * 8);
    bf16x8 o;
#pragma unroll
    for (int j = 0; j < 8; ++j)
      o[j] = (bf16)(__expf((float)v[j] - m) * inv);
    *(bf16x8*)(rowp + ic * 8) = o;
  }
}

// ---------------------------------------------------------------------------
// out = P @ V (f32 out). Pure global-fragment WMMA loop, causal bound.
// ---------------------------------------------------------------------------
__global__ void __launch_bounds__(256)
attn_out(const bf16* __restrict__ P, const bf16* __restrict__ V,
         float* __restrict__ out) {
  const int qTile = blockIdx.x, cTile = blockIdx.y, bz = blockIdx.z;
  const int lane = threadIdx.x & 31, w = threadIdx.x >> 5;
  const int mBase = qTile * 128 + (w & 3) * 32;
  const int nBase = cTile * 128 + (w >> 2) * 64;
  const bf16* Pb = P + (size_t)bz * TLEN * TLEN;
  const bf16* Vb = V + (size_t)bz * TLEN * N_EMBD;
  const int sEnd = qTile * 128 + 128;
  v8f acc[2][4] = {};
  for (int s = 0; s < sEnd; s += 32) {
    v16bf a[2], b[4];
    a[0] = load_a_frag(Pb + (size_t)(mBase)      * TLEN + s, lane, TLEN);
    a[1] = load_a_frag(Pb + (size_t)(mBase + 16) * TLEN + s, lane, TLEN);
#pragma unroll
    for (int j = 0; j < 4; ++j)
      b[j] = load_b_frag(Vb + (size_t)s * N_EMBD + nBase + 16 * j, lane, N_EMBD);
#pragma unroll
    for (int r = 0; r < 2; ++r)
#pragma unroll
      for (int j = 0; j < 4; ++j)
        acc[r][j] = wmma_bf16(a[r], b[j], acc[r][j]);
  }
  float* ob = out + (size_t)bz * TLEN * N_EMBD;
  const int nc = lane & 15, mh = (lane >> 4) << 3;
#pragma unroll
  for (int r = 0; r < 2; ++r)
#pragma unroll
    for (int j = 0; j < 4; ++j)
#pragma unroll
      for (int i = 0; i < 8; ++i) {
        const int m = mBase + 16 * r + mh + i;
        const int n = nBase + 16 * j + nc;
        ob[(size_t)m * N_EMBD + n] = acc[r][j][i];
      }
}

extern "C" void kernel_launch(void* const* d_in, const int* in_sizes, int n_in,
                              void* d_out, int out_size, void* d_ws, size_t ws_size,
                              hipStream_t stream) {
  (void)in_sizes; (void)n_in; (void)out_size; (void)ws_size;
  const float* x  = (const float*)d_in[0];
  const float* Wq = (const float*)d_in[1];
  const float* Wk = (const float*)d_in[2];
  const float* Wv = (const float*)d_in[3];

  char* ws = (char*)d_ws;
  const size_t xBytes   = (size_t)ROWS * N_EMBD * sizeof(bf16);     // 16 MB
  const size_t wBytes   = (size_t)N_EMBD * N_EMBD * sizeof(bf16);   // 2 MB
  bf16* xb  = (bf16*)(ws);
  bf16* Wtq = (bf16*)(ws + xBytes);
  bf16* Wtk = (bf16*)(ws + xBytes + wBytes);
  bf16* Wtv = (bf16*)(ws + xBytes + 2 * wBytes);
  bf16* Qb  = (bf16*)(ws + xBytes + 3 * wBytes);
  bf16* Kt  = (bf16*)(ws + 2 * xBytes + 3 * wBytes);
  bf16* Vb  = (bf16*)(ws + 3 * xBytes + 3 * wBytes);
  bf16* P   = (bf16*)(ws + 4 * xBytes + 3 * wBytes);                // 64 MB

  const float scale = 0.03125f;  // 1/sqrt(1024), folded into Q
  dim3 blk(256);
  cvt_f32_bf16<<<dim3((ROWS * N_EMBD) / 2048), blk, 0, stream>>>(x, xb);
  transpose_cvt_wT<<<dim3(16, 16), blk, 0, stream>>>(Wq, Wtq);
  transpose_cvt_wT<<<dim3(16, 16), blk, 0, stream>>>(Wk, Wtk);
  transpose_cvt_wT<<<dim3(16, 16), blk, 0, stream>>>(Wv, Wtv);
  gemm_bf16_nt<<<dim3(ROWS / 128, N_EMBD / 128), blk, 0, stream>>>(xb, Wtq, Qb, scale, 0);
  gemm_bf16_nt<<<dim3(ROWS / 128, N_EMBD / 128), blk, 0, stream>>>(xb, Wtk, Kt, 1.0f, 1);
  gemm_bf16_nt<<<dim3(ROWS / 128, N_EMBD / 128), blk, 0, stream>>>(xb, Wtv, Vb, 1.0f, 0);
  attn_scores<<<dim3(TLEN / 128, TLEN / 128, BATCH), blk, 0, stream>>>(Qb, Kt, P);
  softmax_rows<<<dim3(ROWS / 8), blk, 0, stream>>>(P);
  attn_out<<<dim3(TLEN / 128, N_EMBD / 128, BATCH), blk, 0, stream>>>(P, Vb, (float*)d_out);
}
